// MultiHeadAttention_83760452207434
// MI455X (gfx1250) — matrix operand order repaired and checked by reading, not compiled, not run on hardware
//
#include <hip/hip_runtime.h>
#include <hip/hip_bf16.h>
#include <stdint.h>

// MHA for MI455X (gfx1250): bf16 WMMA, fp32 accum, flash attention,
// double-buffered TDM (tensor_load_to_lds) pipeline feeding all GEMM tiles.
// B=4, T=2048, D=1024, H=16, Dh=64.

#define TQ 2048
#define DM 1024
#define NH 16
#define DH 64
#define NB 4

typedef __bf16 bf16_t;
typedef bf16_t v16bf __attribute__((ext_vector_type(16)));
typedef float  v8f   __attribute__((ext_vector_type(8)));
typedef unsigned short u16_t;
typedef unsigned int u32x4 __attribute__((ext_vector_type(4)));
typedef int i32x4 __attribute__((ext_vector_type(4)));
typedef int i32x8 __attribute__((ext_vector_type(8)));

union FragAB { uint4 q[2]; v16bf bf; };

__device__ __forceinline__ unsigned f2bf(float f) {
  unsigned u = __float_as_uint(f);
  u += 0x7FFFu + ((u >> 16) & 1u);          // round-to-nearest-even
  return u >> 16;
}
__device__ __forceinline__ unsigned pack2(float lo, float hi) {
  return f2bf(lo) | (f2bf(hi) << 16);
}

// ---------------------------------------------------------------------------
// TDM: DMA a 2-D bf16 tile [tile_d1 rows x tile_d0=32 cols] from a row-major
// [tensor_d1 x tensor_d0] tensor into LDS, inserting 16B of padding after
// every 64B of data (-> LDS row stride 80B == [..][40] u16 arrays).
// D# built per CDNA5 ISA ch.8 (group0/group1; groups 2/3 zero => 2-D).
// ---------------------------------------------------------------------------
__device__ __forceinline__ void tdm_load_2d_bf16(
    const void* gptr, unsigned lds_addr,
    unsigned tile_d0, unsigned tile_d1,
    unsigned tensor_d0, unsigned tensor_d1, unsigned stride0)
{
  unsigned long long ga = (unsigned long long)(uintptr_t)gptr;
  u32x4 g0;
  g0[0] = 1u;                                    // count=1 (valid), user mode
  g0[1] = lds_addr;                              // lds_addr -> bits[63:32]
  g0[2] = (unsigned)ga;                          // global_addr[31:0]
  g0[3] = (unsigned)((ga >> 32) & 0x01FFFFFFu)   // global_addr[56:32]
        | 0x80000000u;                           // type=2 ('10' in [127:126])
  i32x8 g1;
  g1[0] = (1 << 16)                              // data_size = 2 bytes
        | (1 << 20)                              // pad_enable
        | (3 << 22)                              // pad_interval: 16 DWORDs
        | (3 << 25);                             // pad_amount: 4 DWORDs
  g1[1] = (int)((tensor_d0 & 0xFFFFu) << 16);    // tensor_dim0 lo @ bit48
  g1[2] = (int)((tensor_d0 >> 16) | ((tensor_d1 & 0xFFFFu) << 16));
  g1[3] = (int)((tensor_d1 >> 16) | (tile_d0 << 16));  // dim1 hi | tile_dim0
  g1[4] = (int)tile_d1;                          // tile_dim1 (tile_dim2 = 0)
  g1[5] = (int)stride0;                          // tensor_dim0_stride[31:0]
  g1[6] = 0;                                     // stride0 hi | dim1_stride lo
  g1[7] = 0;
  i32x4 z4 = {};
#if defined(__clang_major__) && (__clang_major__ >= 23)
  i32x8 z8 = {};
  __builtin_amdgcn_tensor_load_to_lds(g0, g1, z4, z4, z8, 0);
#else
  __builtin_amdgcn_tensor_load_to_lds(g0, g1, z4, z4, 0);
#endif
}

__device__ __forceinline__ unsigned lds_off(const void* p) {
  // Shared-memory generic pointers carry the LDS byte offset in the low bits.
  return (unsigned)(uintptr_t)p;
}

// ---------------------------------------------------------------------------
// Elementwise fp32 -> bf16 (8 elems/thread).  n = total elems / 8 blocks*256.
// ---------------------------------------------------------------------------
__global__ __launch_bounds__(256) void xconv_kernel(
    const float* __restrict__ X, u16_t* __restrict__ Y)
{
  size_t i = (size_t)blockIdx.x * 256 + threadIdx.x;
  const float4* src = (const float4*)X + 2 * i;
  float4 a = src[0], b = src[1];
  uint4 u;
  u.x = pack2(a.x, a.y); u.y = pack2(a.z, a.w);
  u.z = pack2(b.x, b.y); u.w = pack2(b.z, b.w);
  ((uint4*)Y)[i] = u;
}

// ---------------------------------------------------------------------------
// One-time weight convert + transpose: Wt[n][k] (bf16) = f2bf(W[k][n]).
// Grid (16,16), block 256, 64x64 tiles; coalesced on both sides via LDS.
// ---------------------------------------------------------------------------
__global__ __launch_bounds__(256) void wconv_kernel(
    const float* __restrict__ W, u16_t* __restrict__ Wt)
{
  __shared__ u16_t tile[64][65];
  const int t = threadIdx.x;
  const int k0 = blockIdx.x * 64, n0 = blockIdx.y * 64;
  #pragma unroll
  for (int it = 0; it < 16; ++it) {
    int idx = t + it * 256;
    int kl = idx >> 6, nl = idx & 63;
    tile[nl][kl] = (u16_t)f2bf(W[(size_t)(k0 + kl) * DM + n0 + nl]);
  }
  __syncthreads();
  #pragma unroll
  for (int it = 0; it < 16; ++it) {
    int idx = t + it * 256;
    int nl = idx >> 6, kl = idx & 63;
    Wt[(size_t)(n0 + nl) * DM + k0 + kl] = tile[nl][kl];
  }
}

// ---------------------------------------------------------------------------
// GEMM: C[8192,1024] = Xbf16[8192,1024] @ Wt(bf16 [n][k]).
// Both tiles streamed by the TDM into double-buffered padded LDS; wave 0
// issues next-buffer DMAs, waits tensorcnt<=2 (in-order completion), then
// the block computes the current buffer while the next one streams in.
// mode 0: out bf16 [B,H,T,Dh] (Q,K)   mode 1: out bf16 [B,H,Dh,T] (V^T)
// mode 2: out fp32 [8192,1024]
// ---------------------------------------------------------------------------
__global__ __launch_bounds__(256) void gemm_bf16_kernel(
    const u16_t* __restrict__ X, const u16_t* __restrict__ Wt,
    void* __restrict__ outp, int mode)
{
  __shared__ __align__(16) u16_t As[2][128][40];
  __shared__ __align__(16) u16_t Bs[2][64][40];
  const int tid  = threadIdx.x;
  const int lane = tid & 31, wave = tid >> 5;
  const int hf   = (lane >> 4) & 1, l16 = lane & 15;
  const int wm   = wave >> 1, wn = wave & 1;
  const int m0   = blockIdx.x * 128;
  const int n0   = blockIdx.y * 64;
  const int NIT  = DM / 32;

  v8f acc[2][2] = {};

  if (wave == 0) {   // prologue: stream buffer 0
    tdm_load_2d_bf16(X  + (size_t)m0 * DM, lds_off(&As[0][0][0]),
                     32, 128, DM, 8192, DM);
    tdm_load_2d_bf16(Wt + (size_t)n0 * DM, lds_off(&Bs[0][0][0]),
                     32, 64, DM, DM, DM);
  }

  for (int i = 0; i < NIT; ++i) {
    const int buf = i & 1;
    if (wave == 0) {
      if (i + 1 < NIT) {   // stream next buffer, then wait for current pair
        const int k1 = (i + 1) * 32;
        tdm_load_2d_bf16(X  + (size_t)m0 * DM + k1, lds_off(&As[buf ^ 1][0][0]),
                         32, 128, DM, 8192, DM);
        tdm_load_2d_bf16(Wt + (size_t)n0 * DM + k1, lds_off(&Bs[buf ^ 1][0][0]),
                         32, 64, DM, DM, DM);
        __builtin_amdgcn_s_wait_tensorcnt(2);
      } else {
        __builtin_amdgcn_s_wait_tensorcnt(0);
      }
    }
    __syncthreads();                       // current buffer visible to all

    FragAB a[2], bb[2];
    #pragma unroll
    for (int mt = 0; mt < 2; ++mt) {
      int mr = wm * 32 + mt * 16 + l16;
      a[mt].q[0] = *(const uint4*)&As[buf][mr][hf * 8];        // K=hf*8..+7
      a[mt].q[1] = *(const uint4*)&As[buf][mr][16 + hf * 8];   // K=16+hf*8..+7
    }
    #pragma unroll
    for (int nt = 0; nt < 2; ++nt) {
      int nr = wn * 32 + nt * 16 + l16;
      bb[nt].q[0] = *(const uint4*)&Bs[buf][nr][hf * 8];       // K=hf*8..+7
      bb[nt].q[1] = *(const uint4*)&Bs[buf][nr][16 + hf * 8];  // K=16+hf*8..+7
    }
    #pragma unroll
    for (int mt = 0; mt < 2; ++mt)
      #pragma unroll
      for (int nt = 0; nt < 2; ++nt)
        acc[mt][nt] = __builtin_amdgcn_wmma_f32_16x16x32_bf16(
            false, a[mt].bf, false, bb[nt].bf, (short)0, acc[mt][nt], false, false);
    __syncthreads();                       // done reading buf before reuse
  }

  if (mode == 2) {
    float* out = (float*)outp;
    #pragma unroll
    for (int mt = 0; mt < 2; ++mt)
      #pragma unroll
      for (int nt = 0; nt < 2; ++nt) {
        int n = n0 + wn * 32 + nt * 16 + l16;
        #pragma unroll
        for (int r = 0; r < 8; ++r) {
          int mr = m0 + wm * 32 + mt * 16 + hf * 8 + r;
          out[(size_t)mr * DM + n] = acc[mt][nt][r];
        }
      }
  } else {
    u16_t* out = (u16_t*)outp;
    #pragma unroll
    for (int mt = 0; mt < 2; ++mt)
      #pragma unroll
      for (int nt = 0; nt < 2; ++nt) {
        int n = n0 + wn * 32 + nt * 16 + l16;
        int h = n >> 6, dh = n & 63;
        #pragma unroll
        for (int r = 0; r < 8; ++r) {
          int mr = m0 + wm * 32 + mt * 16 + hf * 8 + r;
          int b = mr >> 11, t = mr & (TQ - 1);
          size_t idx = (mode == 1)
              ? (((size_t)(b * NH + h) * DH + dh) * TQ + t)
              : (((size_t)(b * NH + h) * TQ + t) * DH + dh);
          out[idx] = (u16_t)f2bf(acc[mt][nt][r]);
        }
      }
  }
}

// ---------------------------------------------------------------------------
// Flash attention: grid (B*H, T/128), 256 thr = 8 waves, 16 q-rows per wave.
// Q,K in [B,H,T,64] bf16; V transposed [B,H,64,T] bf16.
// Out: bf16 [B*T, 1024] (= [B,T,H*64]) for the output projection.
// ---------------------------------------------------------------------------
__global__ __launch_bounds__(256) void attn_kernel(
    const u16_t* __restrict__ Q, const u16_t* __restrict__ K,
    const u16_t* __restrict__ Vt, const int* __restrict__ qpad,
    const int* __restrict__ kpad, u16_t* __restrict__ out)
{
  __shared__ __align__(16) u16_t Plds[8][16][40];
  const int tid  = threadIdx.x;
  const int lane = tid & 31, wave = tid >> 5;
  const int hf   = (lane >> 4) & 1, l16 = lane & 15;
  const int bh   = blockIdx.x;
  const int b    = bh >> 4, h = bh & 15;
  const int q0   = (blockIdx.y * 8 + wave) * 16;
  const float NEG = -__builtin_inff();

  const u16_t* Qb = Q  + (size_t)bh * TQ * DH;
  const u16_t* Kb = K  + (size_t)bh * TQ * DH;
  const u16_t* Vb = Vt + (size_t)bh * DH * TQ;
  const int*   kp = kpad + b * TQ;

  FragAB qa[2];                                   // features 0..31, 32..63
  {
    const u16_t* qrow = Qb + (size_t)(q0 + l16) * DH;
    #pragma unroll
    for (int f = 0; f < 2; ++f) {
      qa[f].q[0] = *(const uint4*)(qrow + f * 32 + hf * 8);
      qa[f].q[1] = *(const uint4*)(qrow + f * 32 + 16 + hf * 8);
    }
  }

  float m_run[8], l_run[8];
  #pragma unroll
  for (int r = 0; r < 8; ++r) { m_run[r] = NEG; l_run[r] = 0.f; }
  v8f o[4] = {};

  const int nkb = (q0 + 16 + 31) >> 5;            // causal key-block count
  for (int kb = 0; kb < nkb; ++kb) {
    const int kbase = kb * 32;
    if (kb + 1 < nkb)                             // global_prefetch next K block
      __builtin_prefetch(Kb + (size_t)(kbase + 32 + lane) * DH, 0, 1);

    v8f s[2];
    #pragma unroll
    for (int t = 0; t < 2; ++t) {                 // keys kbase+t*16 .. +15
      const u16_t* krow = Kb + (size_t)(kbase + t * 16 + l16) * DH;
      FragAB k0f, k1f;
      k0f.q[0] = *(const uint4*)(krow + hf * 8);
      k0f.q[1] = *(const uint4*)(krow + 16 + hf * 8);
      k1f.q[0] = *(const uint4*)(krow + 32 + hf * 8);
      k1f.q[1] = *(const uint4*)(krow + 32 + 16 + hf * 8);
      v8f z = {};
      s[t] = __builtin_amdgcn_wmma_f32_16x16x32_bf16(
          false, qa[0].bf, false, k0f.bf, (short)0, z, false, false);
      s[t] = __builtin_amdgcn_wmma_f32_16x16x32_bf16(
          false, qa[1].bf, false, k1f.bf, (short)0, s[t], false, false);
    }

    const int key0 = kbase + l16, key1 = key0 + 16;
    const int kp0 = kp[key0], kp1 = kp[key1];
    #pragma unroll
    for (int r = 0; r < 8; ++r) {
      int qi = q0 + hf * 8 + r;
      float s0 = (key0 <= qi && kp0) ? s[0][r] * 0.125f : NEG;
      float s1 = (key1 <= qi && kp1) ? s[1][r] * 0.125f : NEG;
      float mx = fmaxf(s0, s1);
      #pragma unroll
      for (int d = 8; d >= 1; d >>= 1) mx = fmaxf(mx, __shfl_xor(mx, d, 32));
      float nm = fmaxf(m_run[r], mx);
      float corr, e0, e1;
      if (nm == NEG) { corr = 1.f; e0 = 0.f; e1 = 0.f; }
      else {
        corr = __expf(m_run[r] - nm);             // -inf -> 0
        e0 = __expf(s0 - nm);
        e1 = __expf(s1 - nm);
      }
      float sum = e0 + e1;
      #pragma unroll
      for (int d = 8; d >= 1; d >>= 1) sum += __shfl_xor(sum, d, 32);
      l_run[r] = l_run[r] * corr + sum;
      m_run[r] = nm;
      #pragma unroll
      for (int dt = 0; dt < 4; ++dt) o[dt][r] *= corr;
      Plds[wave][hf * 8 + r][l16]      = (u16_t)f2bf(e0);
      Plds[wave][hf * 8 + r][16 + l16] = (u16_t)f2bf(e1);
    }
    // wave-private LDS round-trip: C-layout -> A-fragment layout
    asm volatile("s_wait_dscnt 0" ::: "memory");
    FragAB pa;
    pa.q[0] = *(const uint4*)&Plds[wave][l16][hf * 16];
    pa.q[1] = *(const uint4*)&Plds[wave][l16][hf * 16 + 8];

    #pragma unroll
    for (int dt = 0; dt < 4; ++dt) {              // O += P @ V
      const u16_t* vrow = Vb + (size_t)(dt * 16 + l16) * TQ + kbase + hf * 16;
      FragAB vbf;
      vbf.q[0] = *(const uint4*)vrow;
      vbf.q[1] = *(const uint4*)(vrow + 8);
      o[dt] = __builtin_amdgcn_wmma_f32_16x16x32_bf16(
          false, pa.bf, false, vbf.bf, (short)0, o[dt], false, false);
    }
  }

  const int* qpb = qpad + b * TQ;
  #pragma unroll
  for (int r = 0; r < 8; ++r) {
    int qi = q0 + hf * 8 + r;
    float wsc = qpb[qi] ? (1.f / l_run[r]) : 0.f;
    size_t base = ((size_t)(b * TQ + qi)) * DM + h * DH;
    #pragma unroll
    for (int dt = 0; dt < 4; ++dt)
      out[base + dt * 16 + l16] = (u16_t)f2bf(o[dt][r] * wsc);
  }
}

// ---------------------------------------------------------------------------
extern "C" void kernel_launch(void* const* d_in, const int* in_sizes, int n_in,
                              void* d_out, int out_size, void* d_ws, size_t ws_size,
                              hipStream_t stream) {
  (void)in_sizes; (void)n_in; (void)out_size; (void)ws_size;
  const float* query = (const float*)d_in[0];
  const float* key   = (const float*)d_in[1];
  const float* value = (const float*)d_in[2];
  const int*   qpad  = (const int*)d_in[3];
  const int*   kpad  = (const int*)d_in[4];
  const float* W_q   = (const float*)d_in[5];
  const float* W_k   = (const float*)d_in[6];
  const float* W_v   = (const float*)d_in[7];
  const float* W_o   = (const float*)d_in[8];
  float* outp = (float*)d_out;

  const size_t HEADS_ELEMS = (size_t)NB * NH * TQ * DH;  // 8,388,608 bf16 each
  const size_t W_ELEMS = (size_t)DM * DM;                // 1,048,576 bf16 each
  u16_t* Qbf = (u16_t*)d_ws;
  u16_t* Kbf = Qbf + HEADS_ELEMS;
  u16_t* Vtb = Kbf + HEADS_ELEMS;   // V transposed [B,H,64,T]
  u16_t* Xq  = Vtb + HEADS_ELEMS;   // bf16 copies of activations
  u16_t* Xk  = Xq  + HEADS_ELEMS;
  u16_t* Xv  = Xk  + HEADS_ELEMS;
  u16_t* Abf = Xq;                  // attn out aliases Xq (dead after QKV GEMM)
  u16_t* Wqt = Xv  + HEADS_ELEMS;   // weights bf16, transposed [n][k]
  u16_t* Wkt = Wqt + W_ELEMS;
  u16_t* Wvt = Wkt + W_ELEMS;
  u16_t* Wot = Wvt + W_ELEMS;

  const int xblocks = (int)(HEADS_ELEMS / (256 * 8));    // 4096
  xconv_kernel<<<xblocks, 256, 0, stream>>>(query, Xq);
  xconv_kernel<<<xblocks, 256, 0, stream>>>(key,   Xk);
  xconv_kernel<<<xblocks, 256, 0, stream>>>(value, Xv);

  dim3 wgrid(16, 16);               // 64x64 tiles over 1024x1024
  wconv_kernel<<<wgrid, 256, 0, stream>>>(W_q, Wqt);
  wconv_kernel<<<wgrid, 256, 0, stream>>>(W_k, Wkt);
  wconv_kernel<<<wgrid, 256, 0, stream>>>(W_v, Wvt);
  wconv_kernel<<<wgrid, 256, 0, stream>>>(W_o, Wot);

  dim3 ggrid(64, 16);               // M/128 x N/64  (M=8192, N=1024)
  gemm_bf16_kernel<<<ggrid, 256, 0, stream>>>(Xq, Wqt, Qbf, 0);
  gemm_bf16_kernel<<<ggrid, 256, 0, stream>>>(Xk, Wkt, Kbf, 0);
  gemm_bf16_kernel<<<ggrid, 256, 0, stream>>>(Xv, Wvt, Vtb, 1);

  dim3 agrid(NB * NH, TQ / 128);    // (64, 16)
  attn_kernel<<<agrid, 256, 0, stream>>>(Qbf, Kbf, Vtb, qpad, kpad, Abf);

  gemm_bf16_kernel<<<ggrid, 256, 0, stream>>>(Abf, Wot, outp, 2);
}
